// HeteroGNN_75093208203805
// MI455X (gfx1250) — matrix operand order repaired and checked by reading, compiled, not run on hardware
//
#include <hip/hip_runtime.h>
#include <hip/hip_bf16.h>

// Problem constants (match reference)
constexpr int kNS  = 20000;
constexpr int kNR  = 100000;
constexpr int kE   = 600000;
constexpr int kD   = 128;
constexpr int kL   = 2;
constexpr int kOut = 2;

typedef __attribute__((ext_vector_type(16))) __bf16 v16bf;
typedef __attribute__((ext_vector_type(8)))  float  v8f;
typedef __attribute__((ext_vector_type(4)))  float  v4f;

union BF16x16 { v16bf v; unsigned short u[16]; };

__device__ __forceinline__ unsigned short f2bf(float f) {
    unsigned int u = __float_as_uint(f);
    unsigned int r = u + 0x7FFFu + ((u >> 16) & 1u);   // round-to-nearest-even
    return (unsigned short)(r >> 16);
}

// ---------------------------------------------------------------- utilities
__global__ void fill_f32(float* __restrict__ p, float v, int n) {
    int i = blockIdx.x * blockDim.x + threadIdx.x;
    if (i < n) p[i] = v;
}

__global__ void copy_f32(float* __restrict__ dst, const float* __restrict__ src, int n) {
    int i = blockIdx.x * blockDim.x + threadIdx.x;
    if (i < n) dst[i] = src[i];
}

// Convert W[k][j] (f32, 128x128) -> Wt[j][k] (bf16) so B fragments are contiguous.
__global__ void prep_wt(unsigned short* __restrict__ wt, const float* __restrict__ w) {
    int i = blockIdx.x * blockDim.x + threadIdx.x;  // 16384
    if (i < kD * kD) {
        int k = i >> 7, j = i & 127;
        wt[j * kD + k] = f2bf(w[i]);
    }
}

__global__ void count_dst(float* __restrict__ cnt, const int* __restrict__ dst, int e) {
    int i = blockIdx.x * blockDim.x + threadIdx.x;
    if (i < e) atomicAdd(&cnt[dst[i]], 1.0f);
}

__global__ void to_rsqrt(float* __restrict__ p, int n) {
    int i = blockIdx.x * blockDim.x + threadIdx.x;
    if (i < n) p[i] = rsqrtf(p[i]);   // deg >= 1 always (self loop)
}

// ------------------------------------------------- edge-parallel aggregation
// 32 threads per edge, float4 per thread: gather x[src] row, atomic-scatter to agg[dst]
__global__ void scatter_sum(float* __restrict__ agg, const float* __restrict__ x,
                            const int* __restrict__ src, const int* __restrict__ dst, int e) {
    int tid = blockIdx.x * blockDim.x + threadIdx.x;
    int ei = tid >> 5;
    int f4 = (tid & 31) << 2;
    if (ei < e) {
        int s = src[ei], d = dst[ei];
        v4f v = *(const v4f*)(x + (size_t)s * kD + f4);
        float* o = agg + (size_t)d * kD + f4;
        atomicAdd(o + 0, v.x); atomicAdd(o + 1, v.y);
        atomicAdd(o + 2, v.z); atomicAdd(o + 3, v.w);
    }
}

__global__ void gcn_scatter(float* __restrict__ out, const float* __restrict__ xw,
                            const float* __restrict__ dis,
                            const int* __restrict__ row, const int* __restrict__ col, int e) {
    int tid = blockIdx.x * blockDim.x + threadIdx.x;
    int ei = tid >> 5;
    int f4 = (tid & 31) << 2;
    if (ei < e) {
        int r = row[ei], c = col[ei];
        float w = dis[r] * dis[c];
        v4f v = *(const v4f*)(xw + (size_t)r * kD + f4);
        float* o = out + (size_t)c * kD + f4;
        atomicAdd(o + 0, w * v.x); atomicAdd(o + 1, w * v.y);
        atomicAdd(o + 2, w * v.z); atomicAdd(o + 3, w * v.w);
    }
}

// self-loop term + bias for GCN: out[i] += dis[n]^2 * xw[i] + b[f]
__global__ void gcn_selfbias(float* __restrict__ out, const float* __restrict__ xw,
                             const float* __restrict__ dis, const float* __restrict__ b, int n) {
    int i = blockIdx.x * blockDim.x + threadIdx.x;
    if (i < n * kD) {
        int node = i >> 7, f = i & 127;
        float di = dis[node];
        out[i] += di * di * xw[i] + b[f];
    }
}

// ------------------------------------------------------------- WMMA GEMM
// out[N x 128] = (A1 * rowscale1) @ W1 + (A2) @ W2 + bias
// W given pre-transposed bf16: Wt[n][k].  One wave per 16-row block,
// full 16x128 output strip held in 8 f32 accumulator fragments.
__global__ __launch_bounds__(256)
void gemm_wmma(float* __restrict__ out,
               const float* __restrict__ A1, const unsigned short* __restrict__ Wt1,
               const float* __restrict__ scale1,
               const float* __restrict__ A2, const unsigned short* __restrict__ Wt2,
               const float* __restrict__ bias, int nrows)
{
    const int lane = threadIdx.x & 31;
    const int wave = threadIdx.x >> 5;
    const int rowblock = blockIdx.x * 8 + wave;          // wave-uniform
    if (rowblock * 16 >= nrows) return;                  // whole wave exits -> EXEC stays full
    const int h    = lane >> 4;                          // half-wave select
    const int ml   = lane & 15;                          // A row within tile
    const int ncol = lane & 15;                          // B/C column within tile
    int arow = rowblock * 16 + ml;
    if (arow >= nrows) arow = nrows - 1;                 // clamp (stores are guarded)

    v8f acc[8];
    #pragma unroll
    for (int t = 0; t < 8; ++t)
        #pragma unroll
        for (int r = 0; r < 8; ++r) acc[t][r] = 0.0f;

    for (int pass = 0; pass < 2; ++pass) {
        const float*          A  = pass ? A2  : A1;
        const unsigned short* Wt = pass ? Wt2 : Wt1;
        const float*          sc = pass ? nullptr : scale1;
        if (!A) continue;                                // uniform branch
        float rs = 1.0f;
        if (sc) rs = 1.0f / fmaxf(sc[arow], 1.0f);       // SAGE mean = agg / max(cnt,1)
        const float* ap = A + (size_t)arow * kD;
        #pragma unroll
        for (int ks = 0; ks < 4; ++ks) {                 // K = 4 steps of 32
            const int k0 = ks * 32 + 16 * h;
            v4f f0 = *(const v4f*)(ap + k0);
            v4f f1 = *(const v4f*)(ap + k0 + 4);
            v4f f2 = *(const v4f*)(ap + k0 + 8);
            v4f f3 = *(const v4f*)(ap + k0 + 12);
            BF16x16 a;
            #pragma unroll
            for (int t = 0; t < 4; ++t) {
                a.u[t]      = f2bf(f0[t] * rs);
                a.u[4 + t]  = f2bf(f1[t] * rs);
                a.u[8 + t]  = f2bf(f2[t] * rs);
                a.u[12 + t] = f2bf(f3[t] * rs);
            }
            const unsigned short* wk = Wt + ks * 32 + h * 16;
            #pragma unroll
            for (int nt = 0; nt < 8; ++nt) {             // 8 output tiles across N=128
                int n = nt * 16 + ncol;
                v16bf b = *(const v16bf*)(const void*)(wk + (size_t)n * kD);
                acc[nt] = __builtin_amdgcn_wmma_f32_16x16x32_bf16(
                    false, a.v, false, b, (short)0, acc[nt], false, false);
            }
        }
    }

    #pragma unroll
    for (int nt = 0; nt < 8; ++nt) {
        float bn = bias ? bias[nt * 16 + ncol] : 0.0f;
        #pragma unroll
        for (int r = 0; r < 8; ++r) {                    // C/D layout: M = r + 8*h
            int m = rowblock * 16 + r + 8 * h;
            if (m < nrows) out[(size_t)m * kD + nt * 16 + ncol] = acc[nt][r] + bn;
        }
    }
}

// --------------------------------------------------------- final 128->2 head
__global__ void linear_out(float* __restrict__ out, const float* __restrict__ x,
                           const float* __restrict__ w, const float* __restrict__ b, int n) {
    int i = blockIdx.x * blockDim.x + threadIdx.x;
    if (i >= n) return;
    const float* xr = x + (size_t)i * kD;
    float a0 = b[0], a1 = b[1];
    #pragma unroll 4
    for (int k = 0; k < kD; ++k) {
        float xv = xr[k];
        a0 += xv * w[k * 2 + 0];
        a1 += xv * w[k * 2 + 1];
    }
    out[i * 2 + 0] = a0;
    out[i * 2 + 1] = a1;
}

// --------------------------------------------------------------------- host
static inline int cdiv(long long a, long long b) { return (int)((a + b - 1) / b); }

extern "C" void kernel_launch(void* const* d_in, const int* in_sizes, int n_in,
                              void* d_out, int out_size, void* d_ws, size_t ws_size,
                              hipStream_t stream) {
    const float* x_subject = (const float*)d_in[0];
    const float* x_region  = (const float*)d_in[1];
    const int*   ei_has    = (const int*)d_in[2];   // [src_s ; dst_r]
    const int*   ei_rev    = (const int*)d_in[3];   // [dst_r ; src_s]
    const int*   ei_fc     = (const int*)d_in[4];
    const float* sr_Wl = (const float*)d_in[5];
    const float* sr_Wr = (const float*)d_in[6];
    const float* sr_b  = (const float*)d_in[7];
    const float* rs_Wl = (const float*)d_in[8];
    const float* rs_Wr = (const float*)d_in[9];
    const float* rs_b  = (const float*)d_in[10];
    const float* gcn_W = (const float*)d_in[11];
    const float* gcn_b = (const float*)d_in[12];
    const float* lsW = (const float*)d_in[13];
    const float* lsb = (const float*)d_in[14];
    const float* lrW = (const float*)d_in[15];
    const float* lrb = (const float*)d_in[16];
    float* out = (float*)d_out;

    size_t off = 0;
    auto alloc = [&](size_t bytes) -> void* {
        void* p = (char*)d_ws + off;
        off += (bytes + 255) & ~(size_t)255;
        return p;
    };
    float* xs[2] = { (float*)alloc((size_t)kNS * kD * 4), (float*)alloc((size_t)kNS * kD * 4) };
    float* xr[2] = { (float*)alloc((size_t)kNR * kD * 4), (float*)alloc((size_t)kNR * kD * 4) };
    float* agg_s = (float*)alloc((size_t)kNS * kD * 4);
    float* agg_r = (float*)alloc((size_t)kNR * kD * 4);   // reused as GCN xw
    float* cnt_s = (float*)alloc((size_t)kNS * 4);
    float* cnt_r = (float*)alloc((size_t)kNR * 4);
    float* dis   = (float*)alloc((size_t)kNR * 4);        // deg -> deg^-1/2
    unsigned short* wt = (unsigned short*)alloc((size_t)5 * kL * kD * kD * 2);
    auto WT = [&](int mat, int l) { return wt + ((size_t)mat * kL + l) * kD * kD; };
    // mat: 0=sr_Wl 1=sr_Wr 2=rs_Wl 3=rs_Wr 4=gcn_W

    const int T = 256;
    const int nSD = kNS * kD, nRD = kNR * kD;

    // weights -> transposed bf16 (once per call)
    for (int l = 0; l < kL; ++l) {
        prep_wt<<<cdiv(kD * kD, T), T, 0, stream>>>(WT(0, l), sr_Wl + (size_t)l * kD * kD);
        prep_wt<<<cdiv(kD * kD, T), T, 0, stream>>>(WT(1, l), sr_Wr + (size_t)l * kD * kD);
        prep_wt<<<cdiv(kD * kD, T), T, 0, stream>>>(WT(2, l), rs_Wl + (size_t)l * kD * kD);
        prep_wt<<<cdiv(kD * kD, T), T, 0, stream>>>(WT(3, l), rs_Wr + (size_t)l * kD * kD);
        prep_wt<<<cdiv(kD * kD, T), T, 0, stream>>>(WT(4, l), gcn_W + (size_t)l * kD * kD);
    }

    // activations into ping-pong buffers
    copy_f32<<<cdiv(nSD, T), T, 0, stream>>>(xs[0], x_subject, nSD);
    copy_f32<<<cdiv(nRD, T), T, 0, stream>>>(xr[0], x_region, nRD);

    // neighbor counts + GCN degrees (fixed across layers)
    fill_f32<<<cdiv(kNS, T), T, 0, stream>>>(cnt_s, 0.0f, kNS);
    fill_f32<<<cdiv(kNR, T), T, 0, stream>>>(cnt_r, 0.0f, kNR);
    fill_f32<<<cdiv(kNR, T), T, 0, stream>>>(dis, 1.0f, kNR);   // self loop
    count_dst<<<cdiv(kE, T), T, 0, stream>>>(cnt_s, ei_rev + kE, kE);
    count_dst<<<cdiv(kE, T), T, 0, stream>>>(cnt_r, ei_has + kE, kE);
    count_dst<<<cdiv(kE, T), T, 0, stream>>>(dis, ei_fc + kE, kE);
    to_rsqrt<<<cdiv(kNR, T), T, 0, stream>>>(dis, kNR);

    const int scatterBlocks = cdiv((long long)kE * 32, T);
    const int gsBlocks = cdiv(kNS / 16, 8);   // 16-row blocks, 8 waves/block
    const int grBlocks = cdiv(kNR / 16, 8);

    int cur = 0;
    for (int l = 0; l < kL; ++l) {
        int nxt = cur ^ 1;
        // SAGE aggregations
        fill_f32<<<cdiv(nSD, T), T, 0, stream>>>(agg_s, 0.0f, nSD);
        scatter_sum<<<scatterBlocks, T, 0, stream>>>(agg_s, xr[cur], ei_rev, ei_rev + kE, kE);
        fill_f32<<<cdiv(nRD, T), T, 0, stream>>>(agg_r, 0.0f, nRD);
        scatter_sum<<<scatterBlocks, T, 0, stream>>>(agg_r, xs[cur], ei_has, ei_has + kE, kE);
        // new_s = mean_s @ rs_Wl + xs @ rs_Wr + rs_b
        gemm_wmma<<<gsBlocks, T, 0, stream>>>(xs[nxt], agg_s, WT(2, l), cnt_s,
                                              xs[cur], WT(3, l), rs_b + (size_t)l * kD, kNS);
        // new_r (SAGE part) = mean_r @ sr_Wl + xr @ sr_Wr + sr_b
        gemm_wmma<<<grBlocks, T, 0, stream>>>(xr[nxt], agg_r, WT(0, l), cnt_r,
                                              xr[cur], WT(1, l), sr_b + (size_t)l * kD, kNR);
        // GCN: xw = xr @ W  (into agg_r, free after the GEMM above)
        gemm_wmma<<<grBlocks, T, 0, stream>>>(agg_r, xr[cur], WT(4, l), nullptr,
                                              nullptr, nullptr, nullptr, kNR);
        gcn_scatter<<<scatterBlocks, T, 0, stream>>>(xr[nxt], agg_r, dis, ei_fc, ei_fc + kE, kE);
        gcn_selfbias<<<cdiv(nRD, T), T, 0, stream>>>(xr[nxt], agg_r, dis, gcn_b + (size_t)l * kD, kNR);
        cur = nxt;
    }

    // output heads (out_subject then out_region, concatenated)
    linear_out<<<cdiv(kNS, T), T, 0, stream>>>(out, xs[cur], lsW, lsb, kNS);
    linear_out<<<cdiv(kNR, T), T, 0, stream>>>(out + (size_t)kNS * kOut, xr[cur], lrW, lrb, kNR);
}
